// Multimodal_Decoder_28355374088296
// MI455X (gfx1250) — compile-verified, matrix operand order checked by reading, not run on hardware
//
#include <hip/hip_runtime.h>
#include <hip/hip_bf16.h>

typedef __attribute__((ext_vector_type(16))) _Float16 v16h;
typedef __attribute__((ext_vector_type(8)))  _Float16 v8h;
typedef __attribute__((ext_vector_type(8)))  float    v8f;

static constexpr int kH = 128;
static constexpr int kE = 16;
static constexpr int kM = 6;
static constexpr int kT = 30;
static constexpr int kD = 2;

union V8F  { v8f  v; float f[8]; };
union V16H { v16h v; _Float16 h[16]; unsigned u[8]; };

__device__ __forceinline__ float fast_sigmoid(float x) {
  float t = __builtin_amdgcn_exp2f(-1.4426950408889634f * x);
  return __builtin_amdgcn_rcpf(1.0f + t);
}
__device__ __forceinline__ float fast_tanh(float x) {
  float t = __builtin_amdgcn_exp2f(2.8853900817779268f * x); // exp(2x)
  return 1.0f - 2.0f * __builtin_amdgcn_rcpf(t + 1.0f);
}
__device__ __forceinline__ float lrelu(float x) { return fmaxf(x, 0.01f * x); }

// LDS layout (dynamic):
//   [0)       W_hh B-frags: [nt=32][kk=4][lane=32][e=16] f16          = 128 KB
//   [131072)  W_ih B-frags full width incl. bias col at k=16:
//             [nt=32][lane=32][e=16] f16                               =  32 KB
//   [163840)  per-wave: hbuf 16x128 f16 (4096) + xbuf 16x16 f16 (512)
//             + relbuf 16x2 f16 (64), stride 4736 bytes, 8 waves
static constexpr int kBfOff   = 0;
static constexpr int kBihOff  = 131072;
static constexpr int kWaveOff = 163840;
static constexpr int kWaveStride = 4736;
static constexpr int kDecSmem = kWaveOff + 8 * kWaveStride; // 201728 B

__global__ __launch_bounds__(256)
void mmdec_lstm_kernel(const float* __restrict__ last_obs_rel,
                       const float* __restrict__ state_h,
                       const float* __restrict__ rand_c,
                       const float* __restrict__ W_se,  const float* __restrict__ b_se,
                       const float* __restrict__ W_ih,  const float* __restrict__ W_hh,
                       const float* __restrict__ b_ih,  const float* __restrict__ b_hh,
                       const float* __restrict__ W_head,const float* __restrict__ b_head,
                       float* __restrict__ traj, int B)
{
  extern __shared__ char smem[];
  _Float16* Bf   = (_Float16*)(smem + kBfOff);
  _Float16* Bih  = (_Float16*)(smem + kBihOff);

  const int tid  = threadIdx.x;
  const int wave = tid >> 5;
  const int lane = tid & 31;
  const int n    = lane & 15;      // C-layout column / A-layout row
  const int half = lane >> 4;      // lane group
  const int m    = blockIdx.y;     // mode
  const int b0   = blockIdx.x * 128 + wave * 16;

  // ---- stage W_hh into f16 B-fragment layout (ISA 16-bit B 32x16) ----
  for (int idx = tid; idx < 32 * 4 * 32 * 16; idx += 256) {
    int e  = idx & 15;
    int ln = (idx >> 4) & 31;
    int kk = (idx >> 9) & 3;
    int nt = idx >> 11;
    int nn = ln & 15, hf = ln >> 4;
    int k  = kk * 32 + ((e < 8 ? e : e + 8) + hf * 8);
    Bf[idx] = (_Float16)W_hh[(nt * 16 + nn) * kH + k];
  }
  // ---- stage W_ih frags; gate bias folded in at k=16 (e=8, half 0) ----
  for (int idx = tid; idx < 32 * 32 * 16; idx += 256) {
    int e  = idx & 15;
    int ln = (idx >> 4) & 31;
    int nt = idx >> 9;
    int nn = ln & 15, hf = ln >> 4;
    float val = 0.f;
    if (e < 8) {
      int k = e + hf * 8;                       // real K: 0..15
      val = W_ih[(nt * 16 + nn) * kE + k];
    } else if (e == 8 && hf == 0) {             // k==16: bias column
      val = b_ih[nt * 16 + nn] + b_hh[nt * 16 + nn];
    }
    Bih[idx] = (_Float16)val;
  }
  __syncthreads();

  _Float16* hbuf   = (_Float16*)(smem + kWaveOff + wave * kWaveStride);
  _Float16* xbuf   = hbuf + 16 * kH;          // +4096 B
  _Float16* relbuf = xbuf + 16 * 16;          // +4608 B

  // ---- init h (LDS, f16 A-source) from state_h ----
  for (int idx = lane; idx < 16 * kH; idx += 32) {
    int row = idx >> 7, k = idx & 127;
    hbuf[row * kH + k] = (_Float16)state_h[(size_t)(b0 + row) * kH + k];
  }
  // ---- init c (registers, C-fragment layout) from rand_c ----
  float c[8][8];
#pragma unroll
  for (int j = 0; j < 8; ++j)
#pragma unroll
    for (int r = 0; r < 8; ++r)
      c[j][r] = rand_c[((size_t)m * B + b0 + r + 8 * half) * kH + j * 16 + n];

  // ---- constant B-fragments held in registers ----
  V16H wse;                    // W_se^T: [2,16]; b_se folded in at k=2
#pragma unroll
  for (int e = 0; e < 16; ++e) wse.h[e] = (_Float16)0.f;
  {
    float w0 = W_se[n * kD + 0], w1 = W_se[n * kD + 1], bs = b_se[n];
    wse.h[0] = half == 0 ? (_Float16)w0 : (_Float16)0.f;
    wse.h[1] = half == 0 ? (_Float16)w1 : (_Float16)0.f;
    wse.h[2] = half == 0 ? (_Float16)bs : (_Float16)0.f;   // bias column k=2
  }
  V16H whf[4];                 // W_head[m]: [128,2] padded to 128x16 (4 frags)
#pragma unroll
  for (int kk = 0; kk < 4; ++kk)
#pragma unroll
    for (int e = 0; e < 16; ++e) {
      int k = kk * 32 + ((e < 8 ? e : e + 8) + half * 8);
      float v = W_head[((size_t)m * kH + k) * kD + (n < 2 ? n : 0)];
      whf[kk].h[e] = (_Float16)(n < 2 ? v : 0.f);
    }
  float bh = b_head[m * kD + (n < 2 ? n : 0)];
  const float bhead = (n < 2) ? bh : 0.f;

  const v8h zero8 = {};
  // A-frag high chunk for x: 1.0 at k=16 (element 8) for half-0 lanes
  v8h xhi = {};
  xhi[0] = (half == 0) ? (_Float16)1.0f : (_Float16)0.f;
  // dword1 of rel A-frag: 1.0 at k=2 (element 2) for half-0 lanes
  const unsigned relpad = (half == 0) ? 0x3C00u : 0u;

  // ---- emb0 = lrelu(last_obs_rel @ W_se^T + b_se) via WMMA ----
  {
    V16H ra;
#pragma unroll
    for (int e = 0; e < 16; ++e) ra.h[e] = (_Float16)0.f;
    float r0 = last_obs_rel[(size_t)(b0 + n) * kD + 0];
    float r1 = last_obs_rel[(size_t)(b0 + n) * kD + 1];
    ra.h[0] = half == 0 ? (_Float16)r0 : (_Float16)0.f;
    ra.h[1] = half == 0 ? (_Float16)r1 : (_Float16)0.f;
    ra.u[1] = relpad;                                      // bias 1.0 at k=2
    v8f acc = {};
    acc = __builtin_amdgcn_wmma_f32_16x16x32_f16(false, ra.v, false, wse.v,
                                                 (short)0, acc, false, false);
    V8F xo; xo.v = acc;
#pragma unroll
    for (int r = 0; r < 8; ++r)
      xbuf[(r + 8 * half) * 16 + n] = (_Float16)lrelu(xo.f[r]);
  }

  auto load_hA = [&](int kt) -> v16h {
    v8h a = *(const v8h*)(hbuf + n * kH + kt * 32 + half * 8);
    v8h b = *(const v8h*)(hbuf + n * kH + kt * 32 + 16 + half * 8);
    return __builtin_shufflevector(a, b, 0,1,2,3,4,5,6,7,8,9,10,11,12,13,14,15);
  };

  v16h hA[4];
#pragma unroll
  for (int kt = 0; kt < 4; ++kt) hA[kt] = load_hA(kt);

#pragma unroll 1
  for (int t = 0; t < kT; ++t) {
    // x A-fragment: 16 real K + 1.0 at k=16 (bias column)
    v8h xlo = *(const v8h*)(xbuf + n * 16 + half * 8);
    v16h xA = __builtin_shufflevector(xlo, xhi,
                                      0,1,2,3,4,5,6,7,8,9,10,11,12,13,14,15);

    // ---- gates = x@W_ih^T + bias + h@W_hh^T; update c, write h to LDS ----
#pragma unroll
    for (int j = 0; j < 8; ++j) {
      V8F acc[4];
#pragma unroll
      for (int gi = 0; gi < 4; ++gi) {
        const int nt = gi * 8 + j;
        v16h bihf = *(const v16h*)(Bih + (nt * 32 + lane) * 16);
        v8f cz = {};
        v8f av = __builtin_amdgcn_wmma_f32_16x16x32_f16(false, xA, false, bihf,
                                                        (short)0, cz, false, false);
#pragma unroll
        for (int kk = 0; kk < 4; ++kk) {
          v16h bf = *(const v16h*)(Bf + ((nt * 4 + kk) * 32 + lane) * 16);
          av = __builtin_amdgcn_wmma_f32_16x16x32_f16(false, hA[kk], false, bf,
                                                      (short)0, av, false, false);
        }
        acc[gi].v = av;
      }
#pragma unroll
      for (int r = 0; r < 8; ++r) {
        float ig = fast_sigmoid(acc[0].f[r]);
        float fg = fast_sigmoid(acc[1].f[r]);
        float gg = fast_tanh(acc[2].f[r]);
        float og = fast_sigmoid(acc[3].f[r]);
        float cn = fg * c[j][r] + ig * gg;
        c[j][r] = cn;
        float hn = og * fast_tanh(cn);
        hbuf[(r + 8 * half) * kH + j * 16 + n] = (_Float16)hn;
      }
    }

    // reload h A-fragments (now h_t)
#pragma unroll
    for (int kt = 0; kt < 4; ++kt) hA[kt] = load_hA(kt);

    // ---- head: rel = h_t @ W_head[m] + b_head ----
    V8F rel;
#pragma unroll
    for (int r = 0; r < 8; ++r) rel.f[r] = bhead;
    v8f rv = rel.v;
#pragma unroll
    for (int kk = 0; kk < 4; ++kk)
      rv = __builtin_amdgcn_wmma_f32_16x16x32_f16(false, hA[kk], false, whf[kk].v,
                                                  (short)0, rv, false, false);
    rel.v = rv;
    if (n < 2) {
#pragma unroll
      for (int r = 0; r < 8; ++r) {
        int row = r + 8 * half;
        traj[(((size_t)(b0 + row)) * kM + m) * (kT * kD) + t * kD + n] = rel.f[r];
        relbuf[row * 2 + n] = (_Float16)rel.f[r];
      }
    }

    // ---- x = lrelu(rel @ W_se^T + b_se) (bias via k=2 column) ----
    {
      V16H ra;
#pragma unroll
      for (int e = 0; e < 16; ++e) ra.h[e] = (_Float16)0.f;
      unsigned p = *(const unsigned*)(relbuf + n * 2);
      ra.u[0] = (half == 0) ? p : 0u;
      ra.u[1] = relpad;
      v8f acc = {};
      acc = __builtin_amdgcn_wmma_f32_16x16x32_f16(false, ra.v, false, wse.v,
                                                   (short)0, acc, false, false);
      V8F xo; xo.v = acc;
#pragma unroll
      for (int r = 0; r < 8; ++r)
        xbuf[(r + 8 * half) * 16 + n] = (_Float16)lrelu(xo.f[r]);
    }
  }
}

// ---------- confidences: 60->60->60 residual MLP -> 1, softmax over modes ----
static constexpr int kTD = 60;
static constexpr int kConfSmem = (3600 * 2 + 180 + 192 * 61 * 2 + 192) * 4;

__global__ __launch_bounds__(192)
void mmdec_conf_kernel(const float* __restrict__ W1, const float* __restrict__ b1,
                       const float* __restrict__ W2, const float* __restrict__ b2,
                       const float* __restrict__ W_out, const float* __restrict__ b_out,
                       const float* __restrict__ traj, float* __restrict__ conf, int B)
{
  extern __shared__ char smem[];
  float* w1  = (float*)smem;           // 3600
  float* w2  = w1 + 3600;              // 3600
  float* bb1 = w2 + 3600;              // 60
  float* bb2 = bb1 + 60;               // 60
  float* wo  = bb2 + 60;               // 60
  float* xs  = wo + 60;                // 192*61 (pad 61: conflict-free)
  float* hs  = xs + 192 * 61;          // 192*61
  float* cs  = hs + 192 * 61;          // 192

  const int tid = threadIdx.x;
  for (int i = tid; i < 3600; i += 192) { w1[i] = W1[i]; w2[i] = W2[i]; }
  for (int i = tid; i < 60;   i += 192) { bb1[i] = b1[i]; bb2[i] = b2[i]; wo[i] = W_out[i]; }

  const int mm = tid % kM, bl = tid / kM;       // 32 batch rows x 6 modes
  const int b  = blockIdx.x * 32 + bl;
  float* x = xs + tid * 61;
  float* h = hs + tid * 61;
  const float* src = traj + ((size_t)b * kM + mm) * kTD;
  for (int k = 0; k < kTD; ++k) x[k] = src[k];
  __syncthreads();

  for (int o = 0; o < kTD; ++o) {
    float s = bb1[o];
    for (int k = 0; k < kTD; ++k) s += x[k] * w1[k * kTD + o];
    h[o] = fmaxf(s, 0.f);
  }
  float cr = b_out[0];
  for (int o = 0; o < kTD; ++o) {
    float s = bb2[o] + x[o];
    for (int k = 0; k < kTD; ++k) s += h[k] * w2[k * kTD + o];
    cr += fmaxf(s, 0.f) * wo[o];
  }
  cs[tid] = cr;
  __syncthreads();

  float mx = -1e30f;
  for (int q = 0; q < kM; ++q) mx = fmaxf(mx, cs[bl * kM + q]);
  float sum = 0.f;
  for (int q = 0; q < kM; ++q)
    sum += __builtin_amdgcn_exp2f(1.4426950408889634f * (cs[bl * kM + q] - mx));
  float e = __builtin_amdgcn_exp2f(1.4426950408889634f * (cr - mx));
  conf[(size_t)b * kM + mm] = e / sum;
}

extern "C" void kernel_launch(void* const* d_in, const int* in_sizes, int n_in,
                              void* d_out, int out_size, void* d_ws, size_t ws_size,
                              hipStream_t stream) {
  const float* last_obs_rel = (const float*)d_in[1];
  const float* state_h      = (const float*)d_in[2];
  const float* rand_c       = (const float*)d_in[3];
  const float* W_se   = (const float*)d_in[4];
  const float* b_se   = (const float*)d_in[5];
  const float* W_ih   = (const float*)d_in[6];
  const float* W_hh   = (const float*)d_in[7];
  const float* b_ih   = (const float*)d_in[8];
  const float* b_hh   = (const float*)d_in[9];
  const float* W_head = (const float*)d_in[10];
  const float* b_head = (const float*)d_in[11];
  const float* W1     = (const float*)d_in[12];
  const float* b1     = (const float*)d_in[13];
  const float* W2     = (const float*)d_in[14];
  const float* b2     = (const float*)d_in[15];
  const float* W_out  = (const float*)d_in[16];
  const float* b_out  = (const float*)d_in[17];

  const int B = in_sizes[0] / kD;                       // 16384
  float* traj = (float*)d_out;
  float* conf = traj + (size_t)B * kM * kT * kD;

  (void)hipFuncSetAttribute((const void*)mmdec_lstm_kernel,
                            hipFuncAttributeMaxDynamicSharedMemorySize, kDecSmem);
  (void)hipFuncSetAttribute((const void*)mmdec_conf_kernel,
                            hipFuncAttributeMaxDynamicSharedMemorySize, kConfSmem);

  dim3 grid(B / 128, kM);
  mmdec_lstm_kernel<<<grid, 256, kDecSmem, stream>>>(
      last_obs_rel, state_h, rand_c, W_se, b_se, W_ih, W_hh, b_ih, b_hh,
      W_head, b_head, traj, B);
  mmdec_conf_kernel<<<B / 32, 192, kConfSmem, stream>>>(
      W1, b1, W2, b2, W_out, b_out, traj, conf, B);
}
